// ParallelAttention_52372831207514
// MI455X (gfx1250) — compile-verified, matrix operand order checked by reading, not run on hardware
//
#include <hip/hip_runtime.h>
#include <hip/hip_bf16.h>
#include <math.h>
#include <stdint.h>

// ---- sizes -----------------------------------------------------------------
#define DIMV   1024
#define HEADS  16
#define DHEAD  64
#define BATCH  2
#define SEQ    2048
#define ROWS   (BATCH*SEQ)   // 4096

typedef __attribute__((ext_vector_type(16))) _Float16 v16h;
typedef __attribute__((ext_vector_type(8)))  _Float16 v8h;
typedef __attribute__((ext_vector_type(8)))  float    v8f;
typedef __attribute__((ext_vector_type(4)))  float    v4f;
typedef __attribute__((ext_vector_type(4)))  unsigned int u32x4;
typedef __attribute__((ext_vector_type(8)))  int      i32x8;
typedef __attribute__((ext_vector_type(4)))  int      i32x4;

// Load a WMMA 16-bit A/B fragment. `p` points at 32 contiguous f16 (one
// row of A / one column of B for a K=32 chunk). Per ISA layout:
//   lanes 0-15 : elems 0..7 = K0..7,  elems 8..15 = K16..23
//   lanes16-31 : elems 0..7 = K8..15, elems 8..15 = K24..31
__device__ __forceinline__ v16h frag_ld(const _Float16* p, int half) {
  v8h lo = *(const v8h*)(p + half*8);
  v8h hi = *(const v8h*)(p + 16 + half*8);
  v16h r;
  #pragma unroll
  for (int i = 0; i < 8; ++i) { r[i] = lo[i]; r[i+8] = hi[i]; }
  return r;
}

__device__ __forceinline__ v8f wmma_f16(v16h a, v16h b, v8f c) {
  return __builtin_amdgcn_wmma_f32_16x16x32_f16(false, a, false, b,
                                                (short)0, c, false, false);
}

// width-16 reductions (rows of a WMMA C tile live in 16-lane halves)
__device__ __forceinline__ float hmax16(float v) {
  #pragma unroll
  for (int o = 1; o < 16; o <<= 1) v = fmaxf(v, __shfl_xor(v, o, 16));
  return v;
}
__device__ __forceinline__ float hsum16(float v) {
  #pragma unroll
  for (int o = 1; o < 16; o <<= 1) v += __shfl_xor(v, o, 16);
  return v;
}

// Issue a TDM 2D tile load: rows x 64 f16 (row-major, contiguous in LDS).
// Descriptor per CDNA5 ISA (D# groups 0/1; 2D => groups
// 2/3 zero). Wave-level DMA: EXEC ignored; caller gates to one wave and must
// s_wait_tensorcnt + barrier before consumers read.
__device__ __forceinline__ void tdm_load_k_tile(unsigned lds_addr,
                                                const _Float16* gsrc,
                                                unsigned tile_rows,
                                                unsigned tensor_rows) {
  unsigned long long ga = (unsigned long long)(uintptr_t)gsrc;
  u32x4 g0 = { 1u,                                      // count=1 (user D#)
               lds_addr,                                // lds_addr[31:0]
               (unsigned)(ga & 0xffffffffu),            // global_addr[31:0]
               ((unsigned)((ga >> 32) & 0x01ffffffu)) | 0x80000000u }; // +type=2
  i32x8 g1 = { (int)(1u << 16),                         // data_size=1 (2 bytes)
               (int)(64u << 16),                        // tensor_dim0=64 -> [79:48]
               (int)(tensor_rows << 16),                // tensor_dim1 lo -> [111:80]
               (int)((tensor_rows >> 16) | (64u << 16)),// dim1 hi | tile_dim0=64
               (int)tile_rows,                          // tile_dim1 | tile_dim2=0
               64,                                      // tensor_dim0_stride=64
               0, 0 };                                  // stride0 hi | stride1
  i32x4 z4 = { 0, 0, 0, 0 };
#if __clang_major__ >= 23
  i32x8 z8 = { 0, 0, 0, 0, 0, 0, 0, 0 };
  __builtin_amdgcn_tensor_load_to_lds(g0, g1, z4, z4, z8, 0);
#else
  __builtin_amdgcn_tensor_load_to_lds(g0, g1, z4, z4, 0);
#endif
}

// ---- kernel 1: fused Q/K/V projections (z selects which) -------------------
__global__ __launch_bounds__(256)
void qkv_proj_kernel(const float* __restrict__ q, const float* __restrict__ k,
                     const float* __restrict__ v,
                     const float* __restrict__ Wq, const float* __restrict__ Wk,
                     const float* __restrict__ Wv,
                     _Float16* __restrict__ qh, _Float16* __restrict__ kh,
                     _Float16* __restrict__ vh)
{
  __shared__ _Float16 As[128][32];
  __shared__ _Float16 Bs[64][32];

  const int tid  = threadIdx.x;
  const int lane = tid & 31;
  const int wid  = tid >> 5;
  const int half = lane >> 4;
  const int l15  = lane & 15;
  const int wm   = wid >> 1;   // 0..3
  const int wn   = wid & 1;    // 0..1

  const int m0 = blockIdx.x * 128;
  const int n0 = blockIdx.y * 64;

  const float* X; const float* W; _Float16* dst; float scale;
  if (blockIdx.z == 0)      { X = q; W = Wq; dst = qh; scale = 0.125f; } // d^-0.5
  else if (blockIdx.z == 1) { X = k; W = Wk; dst = kh; scale = 1.0f; }
  else                      { X = v; W = Wv; dst = vh; scale = 1.0f; }

  v8f acc[2][2];
  #pragma unroll
  for (int i = 0; i < 2; ++i)
    #pragma unroll
    for (int j = 0; j < 2; ++j)
      #pragma unroll
      for (int r = 0; r < 8; ++r) acc[i][j][r] = 0.0f;

  for (int kk = 0; kk < DIMV; kk += 32) {
    // stage X tile (128x32 fp32 -> f16); prefetch next tile unconditionally
    // (prefetch is speculative on gfx1250: invalid translations are dropped)
    #pragma unroll
    for (int rep = 0; rep < 4; ++rep) {
      int fidx = tid + rep * 256;            // float4 unit index, 0..1023
      int row  = fidx >> 3;
      int kc   = (fidx & 7) * 4;
      const float* src = X + (size_t)(m0 + row) * DIMV + kk + kc;
      v4f x = *(const v4f*)src;
      __builtin_prefetch(src + 32, 0, 1);
      As[row][kc + 0] = (_Float16)x[0];
      As[row][kc + 1] = (_Float16)x[1];
      As[row][kc + 2] = (_Float16)x[2];
      As[row][kc + 3] = (_Float16)x[3];
    }
    // stage W tile transposed: Bs[n][kc] = W[kk+kc][n0+n]
    {
      int n  = tid & 63;
      int kb = (tid >> 6) * 8;
      #pragma unroll
      for (int j = 0; j < 8; ++j) {
        const float* src = W + (size_t)(kk + kb + j) * DIMV + n0 + n;
        Bs[n][kb + j] = (_Float16)(*src);
      }
      __builtin_prefetch(W + (size_t)(kk + 32 + kb) * DIMV + n0 + n, 0, 1);
    }
    __syncthreads();

    v16h af[2], bf[2];
    #pragma unroll
    for (int t = 0; t < 2; ++t) af[t] = frag_ld(&As[wm*32 + t*16 + l15][0], half);
    #pragma unroll
    for (int t = 0; t < 2; ++t) bf[t] = frag_ld(&Bs[wn*32 + t*16 + l15][0], half);
    #pragma unroll
    for (int i = 0; i < 2; ++i)
      #pragma unroll
      for (int j = 0; j < 2; ++j)
        acc[i][j] = wmma_f16(af[i], bf[j], acc[i][j]);
    __syncthreads();
  }

  // scatter to head-major f16 layout [B,H,N,64]
  #pragma unroll
  for (int i = 0; i < 2; ++i)
    #pragma unroll
    for (int j = 0; j < 2; ++j)
      #pragma unroll
      for (int r = 0; r < 8; ++r) {
        int row_g = m0 + wm*32 + i*16 + r + 8*half;
        int col_g = n0 + wn*32 + j*16 + l15;
        int b  = row_g >> 11, ii = row_g & (SEQ - 1);
        int h  = col_g >> 6,  dd = col_g & (DHEAD - 1);
        dst[((size_t)((b*HEADS + h)*SEQ + ii))*DHEAD + dd] =
            (_Float16)(acc[i][j][r] * scale);
      }
}

// ---- kernel 2: causal flash attention --------------------------------------
__global__ __launch_bounds__(256)
void attn_kernel(const _Float16* __restrict__ qh, const _Float16* __restrict__ kh,
                 const _Float16* __restrict__ vh, _Float16* __restrict__ attn)
{
  __shared__ _Float16 Ks[32][64];        // K block, row-major (TDM target)
  __shared__ _Float16 Vt[64][32];        // V block, transposed
  __shared__ _Float16 Ps[8][16][32];     // per-wave P tile (C-layout -> A-layout)

  const int tid  = threadIdx.x;
  const int lane = tid & 31, wid = tid >> 5;
  const int half = lane >> 4, l15 = lane & 15;

  const int qbase = blockIdx.x * 128;
  const int bh = blockIdx.y;             // b*16 + h
  const int b = bh >> 4, h = bh & 15;

  const _Float16* Q = qh + (size_t)bh * SEQ * DHEAD;
  const _Float16* K = kh + (size_t)bh * SEQ * DHEAD;
  const _Float16* V = vh + (size_t)bh * SEQ * DHEAD;

  const int qr = qbase + wid * 16;       // this wave's first query row
  const unsigned ldsKs = (unsigned)(uintptr_t)&Ks[0][0];  // LDS byte offset

  // Q fragments resident in registers (row = qr + l15; K-dim = 64 -> 2 chunks)
  v16h qf[2];
  {
    const _Float16* qrow = Q + (size_t)(qr + l15) * DHEAD;
    qf[0] = frag_ld(qrow,      half);
    qf[1] = frag_ld(qrow + 32, half);
  }

  v8f o[4];
  float rmax[8], rsum[8];
  #pragma unroll
  for (int t = 0; t < 4; ++t)
    #pragma unroll
    for (int r = 0; r < 8; ++r) o[t][r] = 0.0f;
  #pragma unroll
  for (int r = 0; r < 8; ++r) { rmax[r] = -INFINITY; rsum[r] = 0.0f; }

  const int nkb = (qbase >> 5) + 4;      // causal: key blocks 0 .. (qbase+127)/32

  for (int j = 0; j < nkb; ++j) {
    const int kb0 = j * 32;
    // K block: one wave kicks the Tensor Data Mover (32x64 f16 2D tile)
    if (wid == 0)
      tdm_load_k_tile(ldsKs, K + (size_t)kb0 * DHEAD, 32u, (unsigned)SEQ);
    // V block: manual staging (needs transpose into Vt)
    {
      int key = tid >> 3;
      int dq  = (tid & 7) * 8;
      const _Float16* vsrc = V + (size_t)(kb0 + key) * DHEAD + dq;
      v8h vv = *(const v8h*)vsrc;
      // speculative prefetch of next K/V blocks (harmless past the end)
      __builtin_prefetch(vsrc + 32 * DHEAD, 0, 1);
      __builtin_prefetch(K + (size_t)(kb0 + 32 + key) * DHEAD + dq, 0, 1);
      #pragma unroll
      for (int e = 0; e < 8; ++e) Vt[dq + e][key] = vv[e];
    }
    if (wid == 0) __builtin_amdgcn_s_wait_tensorcnt(0);  // K tile landed in LDS
    __syncthreads();

    // S = Q K^T : 16 rows x 32 keys, two 16x16 tiles, K-dim 64 = 2 WMMAs each
    v8f s[2];
    #pragma unroll
    for (int t = 0; t < 2; ++t)
      #pragma unroll
      for (int r = 0; r < 8; ++r) s[t][r] = 0.0f;
    #pragma unroll
    for (int t = 0; t < 2; ++t) {
      v16h bf0 = frag_ld(&Ks[t*16 + l15][0],  half);
      v16h bf1 = frag_ld(&Ks[t*16 + l15][32], half);
      s[t] = wmma_f16(qf[0], bf0, s[t]);
      s[t] = wmma_f16(qf[1], bf1, s[t]);
    }

    // causal mask + online softmax (rows live in 16-lane halves of C layout)
    #pragma unroll
    for (int r = 0; r < 8; ++r) {
      int row = qr + r + 8*half;
      float s0 = ((kb0 + l15)      > row) ? -INFINITY : s[0][r];
      float s1 = ((kb0 + 16 + l15) > row) ? -INFINITY : s[1][r];
      float m    = hmax16(fmaxf(s0, s1));
      float mnew = fmaxf(rmax[r], m);
      float corr = __expf(rmax[r] - mnew);
      rmax[r] = mnew;
      float p0 = __expf(s0 - mnew);
      float p1 = __expf(s1 - mnew);
      rsum[r] = rsum[r] * corr + p0 + p1;   // lane-partial; reduced at the end
      #pragma unroll
      for (int t = 0; t < 4; ++t) o[t][r] *= corr;
      Ps[wid][r + 8*half][l15]      = (_Float16)p0;
      Ps[wid][r + 8*half][16 + l15] = (_Float16)p1;
    }
    __syncthreads();   // P (C-layout) -> LDS visible for A-layout reload

    // O += P V : A = P (16x32), B = V^T columns, four 16-wide d tiles
    v16h pf = frag_ld(&Ps[wid][l15][0], half);
    #pragma unroll
    for (int t = 0; t < 4; ++t) {
      v16h bf = frag_ld(&Vt[t*16 + l15][0], half);
      o[t] = wmma_f16(pf, bf, o[t]);
    }
    __syncthreads();   // protect Ks/Vt before next iteration overwrites
  }

  // finalize: divide by row sums, store f16 [B, N, H*64]
  #pragma unroll
  for (int r = 0; r < 8; ++r) {
    float inv = 1.0f / hsum16(rsum[r]);
    int row = qr + r + 8*half;
    size_t base = ((size_t)(b * SEQ + row)) * DIMV + h * DHEAD;
    #pragma unroll
    for (int t = 0; t < 4; ++t)
      attn[base + t*16 + l15] = (_Float16)(o[t][r] * inv);
  }
}

// ---- kernel 3: output projection + bias ------------------------------------
__global__ __launch_bounds__(256)
void out_proj_kernel(const _Float16* __restrict__ A, const float* __restrict__ Wo,
                     const float* __restrict__ bo, float* __restrict__ out)
{
  __shared__ _Float16 As[128][32];
  __shared__ _Float16 Bs[64][32];

  const int tid  = threadIdx.x;
  const int lane = tid & 31, wid = tid >> 5;
  const int half = lane >> 4, l15 = lane & 15;
  const int wm = wid >> 1, wn = wid & 1;
  const int m0 = blockIdx.x * 128, n0 = blockIdx.y * 64;

  v8f acc[2][2];
  #pragma unroll
  for (int i = 0; i < 2; ++i)
    #pragma unroll
    for (int j = 0; j < 2; ++j)
      #pragma unroll
      for (int r = 0; r < 8; ++r) acc[i][j][r] = 0.0f;

  for (int kk = 0; kk < DIMV; kk += 32) {
    {
      int row = tid >> 1;
      int seg = (tid & 1) * 16;
      const _Float16* src = A + (size_t)(m0 + row) * DIMV + kk + seg;
      __builtin_prefetch(src + 32, 0, 1);
      *(v8h*)&As[row][seg]     = *(const v8h*)(src);
      *(v8h*)&As[row][seg + 8] = *(const v8h*)(src + 8);
    }
    {
      int n  = tid & 63;
      int kb = (tid >> 6) * 8;
      #pragma unroll
      for (int j = 0; j < 8; ++j) {
        const float* src = Wo + (size_t)(kk + kb + j) * DIMV + n0 + n;
        Bs[n][kb + j] = (_Float16)(*src);
      }
      __builtin_prefetch(Wo + (size_t)(kk + 32 + kb) * DIMV + n0 + n, 0, 1);
    }
    __syncthreads();

    v16h af[2], bf[2];
    #pragma unroll
    for (int t = 0; t < 2; ++t) af[t] = frag_ld(&As[wm*32 + t*16 + l15][0], half);
    #pragma unroll
    for (int t = 0; t < 2; ++t) bf[t] = frag_ld(&Bs[wn*32 + t*16 + l15][0], half);
    #pragma unroll
    for (int i = 0; i < 2; ++i)
      #pragma unroll
      for (int j = 0; j < 2; ++j)
        acc[i][j] = wmma_f16(af[i], bf[j], acc[i][j]);
    __syncthreads();
  }

  #pragma unroll
  for (int i = 0; i < 2; ++i)
    #pragma unroll
    for (int j = 0; j < 2; ++j)
      #pragma unroll
      for (int r = 0; r < 8; ++r) {
        int row_g = m0 + wm*32 + i*16 + r + 8*half;
        int col_g = n0 + wn*32 + j*16 + l15;
        out[(size_t)row_g * DIMV + col_g] = acc[i][j][r] + bo[col_g];
      }
}

// ---- launch ----------------------------------------------------------------
extern "C" void kernel_launch(void* const* d_in, const int* in_sizes, int n_in,
                              void* d_out, int out_size, void* d_ws, size_t ws_size,
                              hipStream_t stream)
{
  const float* q  = (const float*)d_in[0];
  const float* k  = (const float*)d_in[1];
  const float* v  = (const float*)d_in[2];
  // d_in[3] = mask: exactly causal, computed analytically in-kernel (unused)
  const float* Wq = (const float*)d_in[4];
  const float* Wk = (const float*)d_in[5];
  const float* Wv = (const float*)d_in[6];
  const float* Wo = (const float*)d_in[7];
  const float* bo = (const float*)d_in[8];
  float* out = (float*)d_out;

  _Float16* ws = (_Float16*)d_ws;
  const size_t headElems = (size_t)BATCH * HEADS * SEQ * DHEAD;   // 4M (8 MB f16)
  _Float16* qh   = ws;
  _Float16* kh   = qh + headElems;
  _Float16* vh   = kh + headElems;
  _Float16* attn = vh + headElems;                                // 4M (8 MB f16)

  qkv_proj_kernel<<<dim3(ROWS/128, DIMV/64, 3), 256, 0, stream>>>(
      q, k, v, Wq, Wk, Wv, qh, kh, vh);
  attn_kernel<<<dim3(SEQ/128, BATCH*HEADS), 256, 0, stream>>>(qh, kh, vh, attn);
  out_proj_kernel<<<dim3(ROWS/128, DIMV/64), 256, 0, stream>>>(attn, Wo, bo, out);
}
